// BraidCrossing_81853486727553
// MI455X (gfx1250) — compile-verified, matrix operand order checked by reading, not run on hardware
//
#include <hip/hip_runtime.h>
#include <hip/hip_bf16.h>

// ---------------------------------------------------------------------------
// BraidCrossing on MI455X (gfx1250, wave32, WMMA).
//
// Math note: scale = mean(softmax(logits), -1) == 1/(2*NG) == 1/6 exactly,
// so the W1/gelu/W2/softmax branch contributes nothing -> skipped entirely.
// Heavy op: P = x @ Wp^T + bp (16384x2048x2048), bf16x3 split-precision WMMA
// (hi*hi + hi*lo + lo*hi in fp32 accumulators) for ~fp32 accuracy.
// Split conversion is hoisted to a one-shot pass; the GEMM loop uses
// double-buffered async global->LDS DMA (ASYNCcnt) to overlap with WMMA.
// ---------------------------------------------------------------------------

typedef __bf16 bf16_t;
typedef __attribute__((ext_vector_type(16))) bf16_t          v16bf;
typedef __attribute__((ext_vector_type(8)))  float           v8f;
typedef __attribute__((ext_vector_type(8)))  unsigned short  ushort8;
typedef __attribute__((ext_vector_type(4)))  unsigned short  ushort4v;
typedef __attribute__((ext_vector_type(4)))  int             v4i;

#define LDSK 40   // padded LDS row stride (elements) for 32 K-values; 80B = 5x16B

#if defined(__has_builtin)
#if __has_builtin(__builtin_amdgcn_global_load_async_to_lds_b128) && \
    __has_builtin(__builtin_amdgcn_s_wait_asynccnt)
#define USE_ASYNC_LDS 1
#endif
#endif
#ifndef USE_ASYNC_LDS
#define USE_ASYNC_LDS 0
#endif

__device__ __forceinline__ unsigned short f32_to_bf16_rne(float f) {
    union { float f; unsigned int u; } c; c.f = f;
    unsigned int u = c.u;
    unsigned int r = u + 0x7FFFu + ((u >> 16) & 1u);   // round-to-nearest-even
    return (unsigned short)(r >> 16);
}
__device__ __forceinline__ float bf16_bits_to_f32(unsigned short h) {
    union { unsigned int u; float f; } c; c.u = ((unsigned int)h) << 16;
    return c.f;
}

// one 16-byte global -> LDS chunk
__device__ __forceinline__ void lds_copy16(unsigned short* l, const unsigned short* g) {
#if USE_ASYNC_LDS
    __builtin_amdgcn_global_load_async_to_lds_b128(
        (__attribute__((address_space(1))) v4i*)(g),
        (__attribute__((address_space(3))) v4i*)(l), 0, 0);
#else
    *(ushort8*)l = *(const ushort8*)g;
#endif
}
__device__ __forceinline__ void async_fence() {
#if USE_ASYNC_LDS
    __builtin_amdgcn_s_wait_asynccnt(0);
#endif
}

// ---------------------------------------------------------------------------
// One-shot fp32 -> (bf16 hi, bf16 lo) split.
// ---------------------------------------------------------------------------
__global__ __launch_bounds__(256)
void split_bf16_kernel(const float4* __restrict__ src,
                       ushort4v* __restrict__ hi,
                       ushort4v* __restrict__ lo, int n4)
{
    int idx = blockIdx.x * 256 + threadIdx.x;
    if (idx >= n4) return;
    float4 v = src[idx];
    const float e[4] = { v.x, v.y, v.z, v.w };
    ushort4v h, l;
#pragma unroll
    for (int j = 0; j < 4; ++j) {
        unsigned short hb = f32_to_bf16_rne(e[j]);
        h[j] = hb;
        l[j] = f32_to_bf16_rne(e[j] - bf16_bits_to_f32(hb));
    }
    hi[idx] = h;
    lo[idx] = l;
}

// ---------------------------------------------------------------------------
// GEMM: P[m][n] = sum_k X[m][k] * Wp[n][k] + bp[n]  (bf16x3, fp32 accum)
// Block tile 128x128, 8 wave32s (4Mx2N), wave subtile 32x64, K-step 32.
// Double-buffered LDS, async DMA fills.
// ---------------------------------------------------------------------------
__global__ __launch_bounds__(256)
void braid_mixproj_wmma(const unsigned short* __restrict__ Xhi,
                        const unsigned short* __restrict__ Xlo,
                        const unsigned short* __restrict__ Whi,
                        const unsigned short* __restrict__ Wlo,
                        const float* __restrict__ bp,
                        float* __restrict__ P,
                        int M, int N, int K)
{
    __shared__ unsigned short sm[2][4 * 128 * LDSK];   // 2 x 40 KB

    const int tid   = threadIdx.x;
    const int lane  = tid & 31;
    const int wid   = tid >> 5;
    const int waveM = wid & 3;        // 4 waves along M (32 rows each)
    const int waveN = wid >> 2;       // 2 waves along N (64 cols each)
    const int n0    = blockIdx.x * 128;   // N-tile iterates fastest (L2 reuse)
    const int m0    = blockIdx.y * 128;
    const int half  = lane >> 4;
    const int l15   = lane & 15;

    v8f acc[2][4] = {};

    // tile fill: 4 arrays x (128 rows x 4 chunks of 16B) = 2048 chunks / 256 thr
    auto fill = [&](int buf, int kk) {
        unsigned short* As_hi = sm[buf];
        unsigned short* As_lo = sm[buf] + 128 * LDSK;
        unsigned short* Bs_hi = sm[buf] + 2 * 128 * LDSK;
        unsigned short* Bs_lo = sm[buf] + 3 * 128 * LDSK;
#pragma unroll
        for (int q = 0; q < 2; ++q) {
            int chunk = q * 256 + tid;        // 0..511
            int r = chunk >> 2;               // row 0..127
            int c = chunk & 3;                // 16B chunk in row
            size_t gx = (size_t)(m0 + r) * K + kk + c * 8;
            size_t gw = (size_t)(n0 + r) * K + kk + c * 8;
            int     le = r * LDSK + c * 8;
            lds_copy16(&As_hi[le], &Xhi[gx]);
            lds_copy16(&As_lo[le], &Xlo[gx]);
            lds_copy16(&Bs_hi[le], &Whi[gw]);
            lds_copy16(&Bs_lo[le], &Wlo[gw]);
        }
    };

    int buf = 0;
    fill(0, 0);

    for (int kk = 0; kk < K; kk += 32) {
        async_fence();        // this wave's DMA done
        __syncthreads();      // everyone's DMA done; prior reads of buf^1 retired
        if (kk + 32 < K) fill(buf ^ 1, kk + 32);   // overlap next DMA with WMMA

        const unsigned short* As_hi = sm[buf];
        const unsigned short* As_lo = sm[buf] + 128 * LDSK;
        const unsigned short* Bs_hi = sm[buf] + 2 * 128 * LDSK;
        const unsigned short* Bs_lo = sm[buf] + 3 * 128 * LDSK;

        // fragments (ISA 16-bit A 16x32 striping: half0 -> K 0..7,16..23;
        // half1 -> K 8..15,24..31; two 16B chunks per fragment)
        v16bf a_hi[2], a_lo[2], b_hi[4], b_lo[4];
#pragma unroll
        for (int mi = 0; mi < 2; ++mi) {
            int base = (waveM * 32 + mi * 16 + l15) * LDSK;
            union { v16bf v; ushort8 h[2]; } th, tl;
            th.h[0] = *(const ushort8*)&As_hi[base + half * 8];
            th.h[1] = *(const ushort8*)&As_hi[base + 16 + half * 8];
            tl.h[0] = *(const ushort8*)&As_lo[base + half * 8];
            tl.h[1] = *(const ushort8*)&As_lo[base + 16 + half * 8];
            a_hi[mi] = th.v; a_lo[mi] = tl.v;
        }
#pragma unroll
        for (int ni = 0; ni < 4; ++ni) {
            int base = (waveN * 64 + ni * 16 + l15) * LDSK;
            union { v16bf v; ushort8 h[2]; } th, tl;
            th.h[0] = *(const ushort8*)&Bs_hi[base + half * 8];
            th.h[1] = *(const ushort8*)&Bs_hi[base + 16 + half * 8];
            tl.h[0] = *(const ushort8*)&Bs_lo[base + half * 8];
            tl.h[1] = *(const ushort8*)&Bs_lo[base + 16 + half * 8];
            b_hi[ni] = th.v; b_lo[ni] = tl.v;
        }

        // 3-term split-precision MACs: hi*hi + hi*lo + lo*hi
#pragma unroll
        for (int mi = 0; mi < 2; ++mi)
#pragma unroll
            for (int ni = 0; ni < 4; ++ni) {
                acc[mi][ni] = __builtin_amdgcn_wmma_f32_16x16x32_bf16(
                    false, a_hi[mi], false, b_hi[ni], (short)0, acc[mi][ni], false, false);
                acc[mi][ni] = __builtin_amdgcn_wmma_f32_16x16x32_bf16(
                    false, a_hi[mi], false, b_lo[ni], (short)0, acc[mi][ni], false, false);
                acc[mi][ni] = __builtin_amdgcn_wmma_f32_16x16x32_bf16(
                    false, a_lo[mi], false, b_hi[ni], (short)0, acc[mi][ni], false, false);
            }
        buf ^= 1;
    }

    // epilogue: add bias, store P (C/D layout: VGPR r -> M = r + 8*half)
#pragma unroll
    for (int mi = 0; mi < 2; ++mi)
#pragma unroll
        for (int ni = 0; ni < 4; ++ni) {
            int col   = n0 + waveN * 64 + ni * 16 + l15;
            float bia = bp[col];
#pragma unroll
            for (int r = 0; r < 8; ++r) {
                int row = m0 + waveM * 32 + mi * 16 + half * 8 + r;
                P[(size_t)row * N + col] = acc[mi][ni][r] + bia;
            }
        }
}

// ---------------------------------------------------------------------------
// Fused double-LayerNorm epilogue.  One block per (b,t) row, 256 thr, 8 el/thr.
//   v   = (t==0)   ? x_t : LN(x_t + P_{t-1}/6)
//   out = (t==T-1) ? v   : LN(v + P_{t+1}/6)
// ---------------------------------------------------------------------------
__device__ __forceinline__ float block_reduce_sum(float x, float* sbuf) {
#pragma unroll
    for (int off = 16; off > 0; off >>= 1) x += __shfl_xor(x, off, 32);
    const int wid = threadIdx.x >> 5, lane = threadIdx.x & 31;
    if (lane == 0) sbuf[wid] = x;
    __syncthreads();
    float r = 0.f;
#pragma unroll
    for (int w = 0; w < 8; ++w) r += sbuf[w];
    __syncthreads();
    return r;
}

__device__ __forceinline__ void ln_row(float v[8], const float g[8], const float b[8],
                                       float* sbuf, float invD) {
    float s = 0.f;
#pragma unroll
    for (int i = 0; i < 8; ++i) s += v[i];
    const float mu = block_reduce_sum(s, sbuf) * invD;
    float q = 0.f;
#pragma unroll
    for (int i = 0; i < 8; ++i) { float d = v[i] - mu; q += d * d; }
    const float rstd = rsqrtf(block_reduce_sum(q, sbuf) * invD + 1e-5f);
#pragma unroll
    for (int i = 0; i < 8; ++i) v[i] = (v[i] - mu) * rstd * g[i] + b[i];
}

__global__ __launch_bounds__(256)
void braid_ln_epilogue(const float* __restrict__ X, const float* __restrict__ P,
                       const float* __restrict__ gamma, const float* __restrict__ beta,
                       float* __restrict__ Out, int T, int D)
{
    __shared__ float sbuf[8];
    const int m   = blockIdx.x;          // b*T + t
    const int t   = m & (T - 1);         // T = 2048 (pow2)
    const int tid = threadIdx.x;
    const size_t base = (size_t)m * D;
    const float invD  = 1.0f / (float)D;
    const float SCALE = 1.0f / 6.0f;     // == mean(softmax) over 2*NG=6 lanes

    float v[8], g[8], be[8];
#pragma unroll
    for (int i = 0; i < 8; ++i) {
        int c = tid + i * 256;
        v[i]  = X[base + c];
        g[i]  = gamma[c];
        be[i] = beta[c];
    }
    if (t > 0) {                         // tmp_t = LN(x_t + P_{t-1}/6)
#pragma unroll
        for (int i = 0; i < 8; ++i) v[i] += P[base - D + tid + i * 256] * SCALE;
        ln_row(v, g, be, sbuf, invD);
    }
    if (t < T - 1) {                     // out_t = LN(tmp_t + P_{t+1}/6)
#pragma unroll
        for (int i = 0; i < 8; ++i) v[i] += P[base + D + tid + i * 256] * SCALE;
        ln_row(v, g, be, sbuf, invD);
    }
#pragma unroll
    for (int i = 0; i < 8; ++i) Out[base + tid + i * 256] = v[i];
}

// ---------------------------------------------------------------------------
extern "C" void kernel_launch(void* const* d_in, const int* in_sizes, int n_in,
                              void* d_out, int out_size, void* d_ws, size_t ws_size,
                              hipStream_t stream) {
    // setup_inputs order: x, W1, b1, W2, b2, Wp, bp, gamma, beta
    const float* x     = (const float*)d_in[0];
    const float* Wp    = (const float*)d_in[5];
    const float* bp    = (const float*)d_in[6];
    const float* gamma = (const float*)d_in[7];
    const float* beta  = (const float*)d_in[8];
    // W1/b1/W2/b2 are provably dead: mean(softmax(.)) == 1/6 exactly.

    const int B = 8, T = 2048, D = 2048;
    const int M = B * T;                 // 16384 GEMM rows, K = N = D

    // workspace carve-up (all region sizes are multiples of 4 KB)
    char* ws = (char*)d_ws;
    float* P = (float*)ws;                 ws += (size_t)M * D * sizeof(float);
    unsigned short* Xhi = (unsigned short*)ws; ws += (size_t)M * D * 2;
    unsigned short* Xlo = (unsigned short*)ws; ws += (size_t)M * D * 2;
    unsigned short* Whi = (unsigned short*)ws; ws += (size_t)D * D * 2;
    unsigned short* Wlo = (unsigned short*)ws;

    // 1) one-shot fp32 -> bf16 hi/lo splits
    int n4x = (M * D) / 4, n4w = (D * D) / 4;
    split_bf16_kernel<<<dim3((n4x + 255) / 256), 256, 0, stream>>>(
        (const float4*)x, (ushort4v*)Xhi, (ushort4v*)Xlo, n4x);
    split_bf16_kernel<<<dim3((n4w + 255) / 256), 256, 0, stream>>>(
        (const float4*)Wp, (ushort4v*)Whi, (ushort4v*)Wlo, n4w);

    // 2) WMMA GEMM: P = x @ Wp^T + bp   (N-tiles fastest for L2 reuse)
    dim3 gemmGrid(D / 128, M / 128);     // (16, 128)
    braid_mixproj_wmma<<<gemmGrid, 256, 0, stream>>>(Xhi, Xlo, Whi, Wlo, bp, P, M, D, D);

    // 3) fused double-LayerNorm epilogue
    braid_ln_epilogue<<<dim3(M), 256, 0, stream>>>(x, P, gamma, beta,
                                                   (float*)d_out, T, D);
}
